// QuantumOptimizedLayer_7344394076255
// MI455X (gfx1250) — matrix operand order repaired and checked by reading, not compiled, not run on hardware
//
#include <hip/hip_runtime.h>
#include <hip/hip_bf16.h>

typedef __bf16 bf16_t;
typedef __attribute__((ext_vector_type(16))) __bf16 v16bf;
typedef __attribute__((ext_vector_type(8)))  __bf16 v8bf;
typedef __attribute__((ext_vector_type(4)))  __bf16 v4bf;
typedef __attribute__((ext_vector_type(8)))  float  v8f;

typedef __attribute__((ext_vector_type(4))) unsigned int uint32x4;
typedef __attribute__((ext_vector_type(8))) int          int32x8;
typedef __attribute__((ext_vector_type(4))) int          int32x4;

#define TM 128
#define TN 128
#define TK 32

enum { EPI_TANH = 0, EPI_SCALE = 1, EPI_SILU = 2, EPI_NONE_F32 = 3 };

// Issue a TDM 2D tile load: TM rows x TK bf16 cols, row stride = K elements,
// from global tile start `gaddr` into LDS byte offset `lds_addr`.
// D# layout per CDNA5 ISA (groups 2/3 zero => 2D).
// This toolchain (clang-23/therock) uses the 6-arg builtin form.
__device__ __forceinline__ void tdm_load_tileA(unsigned int lds_addr,
                                               const bf16_t* gaddr,
                                               int M, int K)
{
    unsigned long long ga = (unsigned long long)(size_t)gaddr;
    uint32x4 g0;
    g0[0] = 1u;                                   // count=1, user desc
    g0[1] = lds_addr;                             // lds_addr (bytes)
    g0[2] = (unsigned int)ga;                     // global_addr[31:0]
    g0[3] = (unsigned int)((ga >> 32) & 0x01FFFFFFu) | (2u << 30); // addr[56:32] | type=2

    int32x8 g1;
    g1[0] = 0x00010000;                           // data_size=1 (2 bytes)
    g1[1] = (K & 0xFFFF) << 16;                   // tensor_dim0[15:0]
    g1[2] = ((K >> 16) & 0xFFFF) | ((M & 0xFFFF) << 16);   // dim0[31:16] | dim1[15:0]
    g1[3] = ((M >> 16) & 0xFFFF) | (TK << 16);    // dim1[31:16] | tile_dim0=TK
    g1[4] = TM;                                   // tile_dim1=TM, tile_dim2=0
    g1[5] = K;                                    // tensor_dim0_stride[31:0]
    g1[6] = 0;
    g1[7] = 0;

    int32x4 gz4 = {0, 0, 0, 0};
    int32x8 gz8 = {0, 0, 0, 0, 0, 0, 0, 0};
    __builtin_amdgcn_tensor_load_to_lds(g0, g1, gz4, gz4, gz8, 0);
}

// C = epilogue(A @ W + bias)   A: MxK bf16 row-major, W: KxN bf16 row-major
template <int MODE>
__global__ __launch_bounds__(256)
void gemm_bf16_wmma(const bf16_t* __restrict__ A,
                    const bf16_t* __restrict__ W,
                    const float*  __restrict__ bias,
                    const float*  __restrict__ scale,
                    void*         __restrict__ out,
                    int M, int N, int K)
{
    // double-buffered tiles
    __shared__ bf16_t sA[2][TM * TK];   // row-major: sA[b][row*TK + k]
    __shared__ bf16_t sB[2][TN * TK];   // col-major: sB[b][col*TK + k]

    const int tid  = threadIdx.x;
    const int lane = tid & 31;
    const int wave = tid >> 5;
    const int wm   = wave >> 1;      // 0..3 : 32-row strip
    const int wn   = wave & 1;       // 0..1 : 64-col strip

    const int m0 = blockIdx.y * TM;
    const int n0 = blockIdx.x * TN;

    // B staging: 32 rows x 128 cols, thread -> (r = tid/8, 16-col chunk)
    const int b_r = tid >> 3;
    const int b_c = (tid & 7) << 4;

    const int lh  = (lane < 16) ? 0 : 1;   // lane-half select
    const int l16 = lane & 15;

    const bf16_t* gAtile = A + (size_t)m0 * K;                  // + k0
    const bf16_t* gB = W + (size_t)b_r * N + n0 + b_c;          // + k0*N

    const unsigned int ldsA0 = (unsigned int)(size_t)&sA[0][0];
    const unsigned int ldsA1 = (unsigned int)(size_t)&sA[1][0];

    v8f acc[2][4] = {};

    // ---- preload tile 0 into buffer 0 ----
    if (wave == 0)
        tdm_load_tileA(ldsA0, gAtile, M, K);
    {
        v8bf vb0 = *(const v8bf*)gB;
        v8bf vb1 = *(const v8bf*)(gB + 8);
        #pragma unroll
        for (int j = 0; j < 8; ++j) {
            sB[0][(b_c + j)     * TK + b_r] = vb0[j];
            sB[0][(b_c + 8 + j) * TK + b_r] = vb1[j];
        }
    }
    if (wave == 0)
        __builtin_amdgcn_s_wait_tensorcnt(0);
    __syncthreads();

    const int nk = K / TK;
    for (int kt = 0; kt < nk; ++kt) {
        const int cur = kt & 1;
        const int nxt = cur ^ 1;
        const bool have_next = (kt + 1) < nk;

        // ---- async: TDM pulls next A tile straight into the other LDS buffer ----
        if (have_next && wave == 0)
            tdm_load_tileA(nxt ? ldsA1 : ldsA0, gAtile + (size_t)(kt + 1) * TK, M, K);

        // ---- prefetch next B tile (global -> registers), L2 warm-up 2 ahead ----
        v8bf pb0, pb1;
        if (have_next) {
            const bf16_t* gpB = gB + (size_t)(kt + 1) * TK * N;
            pb0 = *(const v8bf*)gpB;
            pb1 = *(const v8bf*)(gpB + 8);
            if (kt + 2 < nk) {
                __builtin_prefetch(gAtile + (size_t)(kt + 2) * TK, 0, 1);
                __builtin_prefetch(gB + (size_t)(kt + 2) * TK * N, 0, 1);
            }
        }

        // ---- build fragments per documented WMMA VGPR layouts ----
        v16bf afrag[2], bfrag[4];
        #pragma unroll
        for (int mi = 0; mi < 2; ++mi) {
            // A 16x32 bf16: lanes0-15 K=0..7 & 16..23 ; lanes16-31 K=8..15 & 24..31
            const bf16_t* p = &sA[cur][(wm * 32 + mi * 16 + l16) * TK + lh * 8];
            ((uint4*)&afrag[mi])[0] = *(const uint4*)p;
            ((uint4*)&afrag[mi])[1] = *(const uint4*)(p + 16);
        }
        #pragma unroll
        for (int ni = 0; ni < 4; ++ni) {
            // B 32x16 bf16: lane = col, lanes0-15 K=0..7 & 16..23 ; lanes16-31 K=8..15 & 24..31
            const bf16_t* p = &sB[cur][(wn * 64 + ni * 16 + l16) * TK + lh * 8];
            ((uint4*)&bfrag[ni])[0] = *(const uint4*)p;
            ((uint4*)&bfrag[ni])[1] = *(const uint4*)(p + 16);
        }

        #pragma unroll
        for (int mi = 0; mi < 2; ++mi)
            #pragma unroll
            for (int ni = 0; ni < 4; ++ni)
                acc[mi][ni] = __builtin_amdgcn_wmma_f32_16x16x32_bf16(
                    false, afrag[mi], false, bfrag[ni],
                    (short)0, acc[mi][ni], false, false);

        // ---- write prefetched B tile (transposed) into the other buffer ----
        if (have_next) {
            #pragma unroll
            for (int j = 0; j < 8; ++j) {
                sB[nxt][(b_c + j)     * TK + b_r] = pb0[j];
                sB[nxt][(b_c + 8 + j) * TK + b_r] = pb1[j];
            }
        }

        // TDM completion must be visible before the barrier publishes `nxt`
        if (have_next && wave == 0)
            __builtin_amdgcn_s_wait_tensorcnt(0);
        __syncthreads();
    }

    // ---- fused epilogue: bias + activation/scale, store ----
    #pragma unroll
    for (int mi = 0; mi < 2; ++mi) {
        #pragma unroll
        for (int ni = 0; ni < 4; ++ni) {
            const int col = n0 + wn * 64 + ni * 16 + l16;
            const float bv = bias[col];
            const float sc = (MODE == EPI_SCALE) ? scale[col] : 1.0f;
            #pragma unroll
            for (int v = 0; v < 8; ++v) {
                const int row = m0 + wm * 32 + mi * 16 + v + lh * 8;
                float r = acc[mi][ni][v] + bv;
                if (MODE == EPI_TANH)  r = tanhf(r);
                else if (MODE == EPI_SILU)  r = r / (1.0f + __expf(-r));
                else if (MODE == EPI_SCALE) r = r * sc;
                if (MODE == EPI_NONE_F32)
                    ((float*)out)[(size_t)row * N + col] = r;
                else
                    ((bf16_t*)out)[(size_t)row * N + col] = (bf16_t)r;
            }
        }
    }
}

__global__ void convert_f32_bf16(const float* __restrict__ in,
                                 bf16_t* __restrict__ out, int n4)
{
    int i = blockIdx.x * blockDim.x + threadIdx.x;
    if (i < n4) {
        float4 v = *(const float4*)(in + (size_t)i * 4);
        v4bf o;
        o[0] = (bf16_t)v.x; o[1] = (bf16_t)v.y;
        o[2] = (bf16_t)v.z; o[3] = (bf16_t)v.w;
        *(v4bf*)(out + (size_t)i * 4) = o;
    }
}

__global__ void compute_scale(const float* __restrict__ vqc,
                              float* __restrict__ scale, int D)
{
    int d = blockIdx.x * blockDim.x + threadIdx.x;
    if (d < D) {
        float theta = vqc[d * 4 + 0];
        float phi   = vqc[d * 4 + 1];
        scale[d] = __cosf(theta * 0.5f) + __sinf(theta * 0.5f) * __cosf(phi);
    }
}

extern "C" void kernel_launch(void* const* d_in, const int* in_sizes, int n_in,
                              void* d_out, int out_size, void* d_ws, size_t ws_size,
                              hipStream_t stream)
{
    const float* x   = (const float*)d_in[0];
    const float* W1  = (const float*)d_in[1];
    const float* b1  = (const float*)d_in[2];
    const float* W2  = (const float*)d_in[3];
    const float* b2  = (const float*)d_in[4];
    const float* vqc = (const float*)d_in[5];
    const float* W3  = (const float*)d_in[6];
    const float* b3  = (const float*)d_in[7];
    const float* W4  = (const float*)d_in[8];
    const float* b4  = (const float*)d_in[9];

    const int M = 8192;   // B*S = 4*2048
    const int D = 1024;

    char* ws = (char*)d_ws;
    size_t off = 0;
    auto carve = [&](size_t bytes) -> char* {
        char* p = ws + off;
        off += (bytes + 255) & ~(size_t)255;
        return p;
    };
    bf16_t* xb    = (bf16_t*)carve((size_t)M * D * sizeof(bf16_t));
    bf16_t* w1b   = (bf16_t*)carve((size_t)D * D * sizeof(bf16_t));
    bf16_t* w2b   = (bf16_t*)carve((size_t)D * D * sizeof(bf16_t));
    bf16_t* w3b   = (bf16_t*)carve((size_t)D * D * sizeof(bf16_t));
    bf16_t* w4b   = (bf16_t*)carve((size_t)D * D * sizeof(bf16_t));
    bf16_t* hA    = (bf16_t*)carve((size_t)M * D * sizeof(bf16_t));
    bf16_t* hB    = (bf16_t*)carve((size_t)M * D * sizeof(bf16_t));
    float*  scale = (float*)carve((size_t)D * sizeof(float));

    // fp32 -> bf16 conversions
    {
        int n4 = (M * D) / 4;
        convert_f32_bf16<<<(n4 + 255) / 256, 256, 0, stream>>>(x, xb, n4);
        int w4n = (D * D) / 4;
        dim3 g((w4n + 255) / 256);
        convert_f32_bf16<<<g, 256, 0, stream>>>(W1, w1b, w4n);
        convert_f32_bf16<<<g, 256, 0, stream>>>(W2, w2b, w4n);
        convert_f32_bf16<<<g, 256, 0, stream>>>(W3, w3b, w4n);
        convert_f32_bf16<<<g, 256, 0, stream>>>(W4, w4b, w4n);
    }
    compute_scale<<<(D + 255) / 256, 256, 0, stream>>>(vqc, scale, D);

    dim3 grid(D / TN, M / TM);   // (8, 64)
    dim3 block(256);

    // h1 = tanh(x @ W1 + b1)
    gemm_bf16_wmma<EPI_TANH><<<grid, block, 0, stream>>>(xb, w1b, b1, nullptr, hA, M, D, D);
    // q = (h1 @ W2 + b2) * scale
    gemm_bf16_wmma<EPI_SCALE><<<grid, block, 0, stream>>>(hA, w2b, b2, scale, hB, M, D, D);
    // h2 = silu(q @ W3 + b3)
    gemm_bf16_wmma<EPI_SILU><<<grid, block, 0, stream>>>(hB, w3b, b3, nullptr, hA, M, D, D);
    // out = h2 @ W4 + b4   (fp32 output)
    gemm_bf16_wmma<EPI_NONE_F32><<<grid, block, 0, stream>>>(hA, w4b, b4, nullptr, d_out, M, D, D);
}
